// GQAAttention_10479720202315
// MI455X (gfx1250) — compile-verified, matrix operand order checked by reading, not run on hardware
//
#include <hip/hip_runtime.h>
#include <hip/hip_bf16.h>

// ---------------------------------------------------------------------------
// GQA attention block for MI455X (gfx1250, wave32, WMMA bf16 16x16x32 + TDM)
// ---------------------------------------------------------------------------

#define BB 4
#define TT 2048
#define EE 2048
#define NH 16
#define NKV 4
#define HD 128

typedef __attribute__((ext_vector_type(16))) __bf16 bf16x16;
typedef __attribute__((ext_vector_type(8)))  float  f32x8;
typedef __attribute__((ext_vector_type(4)))  unsigned int u32x4;
typedef __attribute__((ext_vector_type(8)))  int i32x8;
typedef __attribute__((ext_vector_type(4)))  int i32x4;

#if __has_builtin(__builtin_amdgcn_tensor_load_to_lds)
#define HAVE_TDM 1
#else
#define HAVE_TDM 0
#endif

__device__ __forceinline__ __bf16 f2bf(float f) {
  unsigned u = __float_as_uint(f);
  u += 0x7FFFu + ((u >> 16) & 1u);          // round-to-nearest-even
  unsigned short s = (unsigned short)(u >> 16);
  return __builtin_bit_cast(__bf16, s);
}

#if HAVE_TDM
// Issue a TDM load of a 2-D bf16 tile: 128 rows x 32 elements (64B rows),
// row stride = strideElems in the source tensor. LDS destination gets 32B of
// padding after every 64B row -> LDS row pitch 96B (= [128][48] bf16 layout).
__device__ __forceinline__ void tdm_load_tile_2d(const void* g, unsigned ldsOff,
                                                 int strideElems) {
  unsigned long long ga = (unsigned long long)g;
  u32x4 g0;
  g0.x = 1u;                                    // count=1 (valid user D#)
  g0.y = ldsOff;                                // lds_addr (bytes)
  g0.z = (unsigned)(ga & 0xffffffffu);          // global_addr[31:0]
  g0.w = (unsigned)((ga >> 32) & 0x01ffffffu)   // global_addr[56:32]
       | (2u << 30);                            // type = 2 ("image")
  i32x8 g1;
  g1[0] = (1 << 16)       // data_size = 2 bytes
        | (1 << 20)       // pad_enable
        | (3 << 22)       // pad_interval: every 16 DWORDs (64B)
        | (7 << 25);      // pad_amount: 8 DWORDs (32B)
  g1[1] = (int)(32u << 16);                 // tensor_dim0 = 32 (lo16 @ [63:48])
  g1[2] = (int)(128u << 16);                // dim0 hi16=0 | tensor_dim1 = 128
  g1[3] = (int)(32u << 16);                 // dim1 hi16=0 | tile_dim0 = 32
  g1[4] = 128;                              // tile_dim1 = 128 | tile_dim2 = 0
  g1[5] = strideElems;                      // tensor_dim0_stride[31:0]
  g1[6] = 0;                                // stride0 hi | stride1 lo
  g1[7] = 0;                                // stride1 hi
  i32x4 z4 = {0, 0, 0, 0};                  // groups 2/3 unused (2-D tensor)
  i32x8 z8 = {0, 0, 0, 0, 0, 0, 0, 0};      // extra group (6-arg toolchain)
  __builtin_amdgcn_tensor_load_to_lds(g0, g1, z4, z4, z8, 0);
}
#endif

// ---------------------------------------------------------------------------
// elementwise converts / transposes
// ---------------------------------------------------------------------------
__global__ void conv_f32_bf16(const float* __restrict__ src,
                              __bf16* __restrict__ dst, size_t n) {
  size_t i = (size_t)blockIdx.x * blockDim.x + threadIdx.x;
  if (i < n) dst[i] = f2bf(src[i]);
}

// w[K][N] f32 -> wt[N][K] bf16 (so B-fragments become contiguous loads)
__global__ void transpose_w(const float* __restrict__ w,
                            __bf16* __restrict__ wt, int K, int N) {
  size_t i = (size_t)blockIdx.x * blockDim.x + threadIdx.x;
  if (i >= (size_t)K * N) return;
  int k = (int)(i % K);
  int n = (int)(i / K);
  wt[i] = f2bf(w[(size_t)k * N + n]);
}

// v_raw[B][T][NKV*HD] f32 -> vt[B][NKV][HD][T] bf16
__global__ void transpose_v(const float* __restrict__ v,
                            __bf16* __restrict__ vt) {
  size_t i = (size_t)blockIdx.x * blockDim.x + threadIdx.x;
  size_t total = (size_t)BB * NKV * HD * TT;
  if (i >= total) return;
  size_t r = i;
  int t  = (int)(r % TT); r /= TT;
  int d  = (int)(r % HD); r /= HD;
  int kv = (int)(r % NKV);
  int b  = (int)(r / NKV);
  vt[i] = f2bf(v[((size_t)b * TT + t) * (NKV * HD) + (size_t)kv * HD + d]);
}

// src[B][T][nh*HD] f32 -> dst[B][nh][T][HD] bf16 with RoPE
__global__ void rope_kernel(const float* __restrict__ src,
                            const float* __restrict__ cosA,
                            const float* __restrict__ sinA,
                            __bf16* __restrict__ dst, int nh) {
  size_t i = (size_t)blockIdx.x * blockDim.x + threadIdx.x;
  size_t total = (size_t)BB * nh * TT * (HD / 2);
  if (i >= total) return;
  size_t r = i;
  int p = (int)(r % (HD / 2)); r /= (HD / 2);
  int t = (int)(r % TT);       r /= TT;
  int h = (int)(r % nh);
  int b = (int)(r / nh);
  float c = cosA[(size_t)t * (HD / 2) + p];
  float s = sinA[(size_t)t * (HD / 2) + p];
  size_t ib = ((size_t)b * TT + t) * ((size_t)nh * HD) + (size_t)h * HD + 2 * p;
  float x1 = src[ib], x2 = src[ib + 1];
  size_t ob = (((size_t)b * nh + h) * TT + t) * HD + 2 * p;
  dst[ob]     = f2bf(x1 * c - x2 * s);
  dst[ob + 1] = f2bf(x1 * s + x2 * c);
}

// ---------------------------------------------------------------------------
// bf16 GEMM: C[M][N] = A[M][K] * Bt[N][K]^T   (Bt is B transposed, row-major)
// 128x128 block tile, 8 waves (4 row x 2 col), each wave 32x64 (2x4 WMMA tiles)
// Tiles staged into LDS by the Tensor Data Mover, double-buffered on TENSORcnt.
// ---------------------------------------------------------------------------
__global__ __launch_bounds__(256)
void gemm_bf16(const __bf16* __restrict__ A, const __bf16* __restrict__ Bt,
               float* __restrict__ C, int M, int N, int K) {
  __shared__ __attribute__((aligned(128))) __bf16 As[2][128][48];
  __shared__ __attribute__((aligned(128))) __bf16 Bs[2][128][48];

  const int m0 = blockIdx.y * 128, n0 = blockIdx.x * 128;
  const int tid = threadIdx.x, wave = tid >> 5, lane = tid & 31;
  const int half = lane >> 4, col = lane & 15;
  const int wr = (wave & 3) * 32;   // wave row offset in tile
  const int wc = (wave >> 2) * 64;  // wave col offset in tile

  f32x8 acc[2][4] = {};
  const int nIter = K / 32;

#if HAVE_TDM
  const unsigned ldsA = (unsigned)(unsigned long long)&As[0][0][0];
  const unsigned ldsB = (unsigned)(unsigned long long)&Bs[0][0][0];
  const unsigned bufBytes = 128u * 48u * 2u;  // 12 KB per buffer

  if (wave == 0) {  // TDM ignores EXEC; one issue per workgroup
    tdm_load_tile_2d(A  + (size_t)m0 * K, ldsA, K);
    tdm_load_tile_2d(Bt + (size_t)n0 * K, ldsB, K);
  }
  for (int it = 0; it < nIter; ++it) {
    const int buf = it & 1;
    if (wave == 0) {
      if (it + 1 < nIter) {
        const unsigned nb = (unsigned)((it + 1) & 1) * bufBytes;
        tdm_load_tile_2d(A  + (size_t)m0 * K + (it + 1) * 32, ldsA + nb, K);
        tdm_load_tile_2d(Bt + (size_t)n0 * K + (it + 1) * 32, ldsB + nb, K);
        __builtin_amdgcn_s_wait_tensorcnt(2);  // tile `it` done (in-order)
      } else {
        __builtin_amdgcn_s_wait_tensorcnt(0);
      }
    }
    __syncthreads();  // publish tile `it` to all waves

    bf16x16 af[2], bfr[4];
#pragma unroll
    for (int i = 0; i < 2; i++)
      af[i] = *(const bf16x16*)(&As[buf][wr + i * 16 + col][half * 16]);
#pragma unroll
    for (int j = 0; j < 4; j++)
      bfr[j] = *(const bf16x16*)(&Bs[buf][wc + j * 16 + col][half * 16]);
#pragma unroll
    for (int i = 0; i < 2; i++)
#pragma unroll
      for (int j = 0; j < 4; j++)
        acc[i][j] = __builtin_amdgcn_wmma_f32_16x16x32_bf16(
            false, af[i], false, bfr[j], (short)0, acc[i][j], false, false);

    __syncthreads();  // readers done before this buffer is overwritten
  }
#else
  const int lrow = tid >> 1, lhalf = tid & 1;  // manual fallback path
  for (int k0 = 0; k0 < K; k0 += 32) {
    bf16x16 av = *(const bf16x16*)(A  + (size_t)(m0 + lrow) * K + k0 + lhalf * 16);
    bf16x16 bv = *(const bf16x16*)(Bt + (size_t)(n0 + lrow) * K + k0 + lhalf * 16);
    __syncthreads();
    *(bf16x16*)(&As[0][lrow][lhalf * 16]) = av;
    *(bf16x16*)(&Bs[0][lrow][lhalf * 16]) = bv;
    __syncthreads();
    bf16x16 af[2], bfr[4];
#pragma unroll
    for (int i = 0; i < 2; i++)
      af[i] = *(const bf16x16*)(&As[0][wr + i * 16 + col][half * 16]);
#pragma unroll
    for (int j = 0; j < 4; j++)
      bfr[j] = *(const bf16x16*)(&Bs[0][wc + j * 16 + col][half * 16]);
#pragma unroll
    for (int i = 0; i < 2; i++)
#pragma unroll
      for (int j = 0; j < 4; j++)
        acc[i][j] = __builtin_amdgcn_wmma_f32_16x16x32_bf16(
            false, af[i], false, bfr[j], (short)0, acc[i][j], false, false);
  }
#endif

#pragma unroll
  for (int i = 0; i < 2; i++)
#pragma unroll
    for (int j = 0; j < 4; j++)
#pragma unroll
      for (int v = 0; v < 8; v++) {
        int r = m0 + wr + i * 16 + v + half * 8;
        int c = n0 + wc + j * 16 + col;
        C[(size_t)r * N + c] = acc[i][j][v];
      }
}

// ---------------------------------------------------------------------------
// Flash attention (causal, GQA). Q[B][NH][T][HD], K[B][NKV][T][HD],
// Vt[B][NKV][HD][T] (transposed), Y[B][T][NH*HD] bf16.
// Block: 128 query rows, 8 waves x 16 rows. Key blocks of 64.
// ---------------------------------------------------------------------------
__global__ __launch_bounds__(256)
void attn_kernel(const __bf16* __restrict__ Q, const __bf16* __restrict__ Km,
                 const __bf16* __restrict__ Vt, __bf16* __restrict__ Y) {
  __shared__ __attribute__((aligned(64))) __bf16 Pl[8][16][80];

  const int bh = blockIdx.y;
  const int b = bh / NH, h = bh % NH, kv = h / (NH / NKV);
  const int q0 = blockIdx.x * 128;
  const int wave = threadIdx.x >> 5, lane = threadIdx.x & 31;
  const int half = lane >> 4, col = lane & 15;
  const int qrow = q0 + wave * 16;

  const __bf16* qb = Q  + (((size_t)b * NH  + h ) * TT + qrow) * HD;
  const __bf16* kb = Km + (((size_t)b * NKV + kv) * TT) * HD;
  const __bf16* vb = Vt + (((size_t)b * NKV + kv) * HD) * TT;

  bf16x16 aq[4];
#pragma unroll
  for (int kd = 0; kd < 4; kd++)
    aq[kd] = *(const bf16x16*)(qb + (size_t)col * HD + kd * 32 + half * 16);

  f32x8 o[8] = {};
  float mI[8], lI[8];
#pragma unroll
  for (int v = 0; v < 8; v++) { mI[v] = -1e30f; lI[v] = 0.f; }
  const float scale = 0.08838834764831845f;  // 1/sqrt(128)

  for (int j0 = 0; j0 < qrow + 16; j0 += 64) {
    // prefetch next key block while this one computes (global_prefetch_b8)
    if (j0 + 64 < qrow + 16) {
      __builtin_prefetch(kb + (size_t)(j0 + 64 + lane) * HD, 0, 1);
      __builtin_prefetch(kb + (size_t)(j0 + 96 + lane) * HD, 0, 1);
    }

    // ---- S = Q K^T (16 x 64) ----
    f32x8 s[4] = {};
#pragma unroll
    for (int kd = 0; kd < 4; kd++)
#pragma unroll
      for (int jt = 0; jt < 4; jt++) {
        bf16x16 kf = *(const bf16x16*)(kb + (size_t)(j0 + jt * 16 + col) * HD +
                                       kd * 32 + half * 16);
        s[jt] = __builtin_amdgcn_wmma_f32_16x16x32_bf16(
            false, aq[kd], false, kf, (short)0, s[jt], false, false);
      }

    // ---- scale + causal mask ----
#pragma unroll
    for (int jt = 0; jt < 4; jt++)
#pragma unroll
      for (int v = 0; v < 8; v++) {
        int key = j0 + jt * 16 + col;
        int qr  = qrow + v + half * 8;
        float val = s[jt][v] * scale;
        s[jt][v] = (key <= qr) ? val : -1e30f;
      }

    // ---- online softmax (row reductions across 16-lane half-wave) ----
#pragma unroll
    for (int v = 0; v < 8; v++) {
      float mx = fmaxf(fmaxf(s[0][v], s[1][v]), fmaxf(s[2][v], s[3][v]));
#pragma unroll
      for (int off = 8; off >= 1; off >>= 1)
        mx = fmaxf(mx, __shfl_xor(mx, off, 32));
      float mNew  = fmaxf(mI[v], mx);
      float alpha = __expf(mI[v] - mNew);
      float rs = 0.f;
#pragma unroll
      for (int jt = 0; jt < 4; jt++) {
        float p = __expf(s[jt][v] - mNew);
        s[jt][v] = p;
        rs += p;
      }
#pragma unroll
      for (int off = 8; off >= 1; off >>= 1)
        rs += __shfl_xor(rs, off, 32);
      lI[v] = lI[v] * alpha + rs;
      mI[v] = mNew;
#pragma unroll
      for (int dt = 0; dt < 8; dt++) o[dt][v] *= alpha;
    }

    // ---- P -> per-wave LDS (C-layout -> A-layout transit) ----
#pragma unroll
    for (int jt = 0; jt < 4; jt++)
#pragma unroll
      for (int v = 0; v < 8; v++)
        Pl[wave][v + half * 8][jt * 16 + col] = f2bf(s[jt][v]);
    asm volatile("s_wait_dscnt 0" ::: "memory");

    // ---- O += P V (16 x 128) ----
#pragma unroll
    for (int kc = 0; kc < 2; kc++) {
      bf16x16 pf = *(const bf16x16*)(&Pl[wave][col][kc * 32 + half * 16]);
#pragma unroll
      for (int dt = 0; dt < 8; dt++) {
        bf16x16 vf = *(const bf16x16*)(vb + (size_t)(dt * 16 + col) * TT + j0 +
                                       kc * 32 + half * 16);
        o[dt] = __builtin_amdgcn_wmma_f32_16x16x32_bf16(
            false, pf, false, vf, (short)0, o[dt], false, false);
      }
    }
  }

  // ---- normalize and write Y[b][t][h*HD + d] ----
#pragma unroll
  for (int v = 0; v < 8; v++) {
    float inv = 1.0f / lI[v];
    int tq = qrow + v + half * 8;
    __bf16* yb = Y + ((size_t)b * TT + tq) * ((size_t)NH * HD) + (size_t)h * HD;
#pragma unroll
    for (int dt = 0; dt < 8; dt++)
      yb[dt * 16 + col] = f2bf(o[dt][v] * inv);
  }
}

// ---------------------------------------------------------------------------
// launcher
// ---------------------------------------------------------------------------
extern "C" void kernel_launch(void* const* d_in, const int* in_sizes, int n_in,
                              void* d_out, int out_size, void* d_ws, size_t ws_size,
                              hipStream_t stream) {
  const float* x    = (const float*)d_in[0];
  const float* cosA = (const float*)d_in[1];
  const float* sinA = (const float*)d_in[2];
  const float* w_q  = (const float*)d_in[3];
  const float* w_k  = (const float*)d_in[4];
  const float* w_v  = (const float*)d_in[5];
  const float* w_o  = (const float*)d_in[6];
  float* out = (float*)d_out;

  const size_t BT = (size_t)BB * TT;           // 8192
  char* ws = (char*)d_ws;
  size_t off = 0;
  auto alloc = [&](size_t bytes) -> void* {
    void* p = ws + off;
    off += (bytes + 255) & ~(size_t)255;
    return p;
  };

  __bf16* xb   = (__bf16*)alloc(BT * EE * 2);
  __bf16* wqT  = (__bf16*)alloc((size_t)EE * (NH  * HD) * 2);
  __bf16* wkT  = (__bf16*)alloc((size_t)EE * (NKV * HD) * 2);
  __bf16* wvT  = (__bf16*)alloc((size_t)EE * (NKV * HD) * 2);
  __bf16* woT  = (__bf16*)alloc((size_t)(NH * HD) * EE * 2);
  float*  qraw = (float*)alloc(BT * (NH  * HD) * 4);
  float*  kraw = (float*)alloc(BT * (NKV * HD) * 4);
  float*  vraw = (float*)alloc(BT * (NKV * HD) * 4);
  __bf16* qbf  = (__bf16*)alloc(BT * (NH  * HD) * 2);
  __bf16* kbf  = (__bf16*)alloc(BT * (NKV * HD) * 2);
  __bf16* vtb  = (__bf16*)alloc(BT * (NKV * HD) * 2);
  __bf16* ybf  = (__bf16*)alloc(BT * (NH  * HD) * 2);

  auto blocks = [](size_t n, int t) { return (unsigned)((n + t - 1) / t); };

  // 1. convert inputs to bf16 (weights transposed to [N][K])
  conv_f32_bf16<<<blocks(BT * EE, 256), 256, 0, stream>>>(x, xb, BT * EE);
  transpose_w<<<blocks((size_t)EE * NH * HD, 256), 256, 0, stream>>>(w_q, wqT, EE, NH * HD);
  transpose_w<<<blocks((size_t)EE * NKV * HD, 256), 256, 0, stream>>>(w_k, wkT, EE, NKV * HD);
  transpose_w<<<blocks((size_t)EE * NKV * HD, 256), 256, 0, stream>>>(w_v, wvT, EE, NKV * HD);
  transpose_w<<<blocks((size_t)(NH * HD) * EE, 256), 256, 0, stream>>>(w_o, woT, NH * HD, EE);

  // 2. QKV projections (TDM double-buffered WMMA GEMM)
  gemm_bf16<<<dim3((NH * HD) / 128, BT / 128), 256, 0, stream>>>(xb, wqT, qraw,
      (int)BT, NH * HD, EE);
  gemm_bf16<<<dim3((NKV * HD) / 128, BT / 128), 256, 0, stream>>>(xb, wkT, kraw,
      (int)BT, NKV * HD, EE);
  gemm_bf16<<<dim3((NKV * HD) / 128, BT / 128), 256, 0, stream>>>(xb, wvT, vraw,
      (int)BT, NKV * HD, EE);

  // 3. RoPE (Q,K) + V transpose, all to bf16 head-major layouts
  rope_kernel<<<blocks((size_t)BB * NH * TT * (HD / 2), 256), 256, 0, stream>>>(
      qraw, cosA, sinA, qbf, NH);
  rope_kernel<<<blocks((size_t)BB * NKV * TT * (HD / 2), 256), 256, 0, stream>>>(
      kraw, cosA, sinA, kbf, NKV);
  transpose_v<<<blocks((size_t)BB * NKV * HD * TT, 256), 256, 0, stream>>>(vraw, vtb);

  // 4. causal GQA flash attention
  attn_kernel<<<dim3(TT / 128, BB * NH), 256, 0, stream>>>(qbf, kbf, vtb, ybf);

  // 5. output projection -> f32 d_out
  gemm_bf16<<<dim3(EE / 128, BT / 128), 256, 0, stream>>>(ybf, woT, out,
      (int)BT, EE, NH * HD);
}
